// SelfAttention_51539608364
// MI455X (gfx1250) — matrix operand order repaired and checked by reading, not compiled, not run on hardware
//
#include <hip/hip_runtime.h>
#include <hip/hip_bf16.h>

// ---------------------------------------------------------------------------
// Self-attention (SAGAN style), B=4, N=4096, C=256, Ck=32, fp32 in/out.
// All GEMMs on v_wmma_f32_16x16x32_bf16 (fp32 accumulate). Flash-style fused
// softmax(G F^T) H with online max — S (268 MB) never materialized.
//   k0: repack W into exact WMMA-B-operand order (one-time, tiny)
//   k1: projections f/g (bf16 [B*N,32]) and h transposed hT (bf16 [B][C][N])
//   k2: fused flash attention + gamma/residual
// ---------------------------------------------------------------------------

typedef __attribute__((ext_vector_type(16))) __bf16 v16bf;
typedef __attribute__((ext_vector_type(8)))  __bf16 v8bf;
typedef __attribute__((ext_vector_type(8)))  float  v8f;

#define BB 4
#define NN 4096
#define CC 256
#define CK 32
#define NT 20          // output n-tiles: 2 (f) + 2 (g) + 16 (h)

static __device__ inline v16bf cat8(v8bf lo, v8bf hi) {
    return __builtin_shufflevector(lo, hi, 0,1,2,3,4,5,6,7,8,9,10,11,12,13,14,15);
}

// ---------------------------------------------------------------------------
// Kernel 0: repack Wf|Wg|Wh into WMMA B-operand layout, bf16.
// Wpack[nt][kstep][lane][j]  (nt<2: Wf, nt<4: Wg, else Wh)
//   lane<16 : col = c0+lane,    K = kstep*32 + j        (j = 0..15)
//   lane>=16: col = c0+lane-16, K = kstep*32 + 16 + j
// 20*8*32*16 = 81920 bf16 = 160 KB; loaded contiguously (32B/lane) by k1.
// ---------------------------------------------------------------------------
__global__ __launch_bounds__(256)
void repack_kernel(const float* __restrict__ Wf, const float* __restrict__ Wg,
                   const float* __restrict__ Wh, __bf16* __restrict__ Wpack)
{
    const int idx   = blockIdx.x * 256 + threadIdx.x;   // 320 blocks -> 81920
    const int j     = idx & 15;
    const int lane  = (idx >> 4) & 31;
    const int kstep = (idx >> 9) & 7;
    const int nt    = idx >> 12;                        // 0..19

    const float* W;  int ldW;
    if (nt < 2)      { W = Wf; ldW = CK; }
    else if (nt < 4) { W = Wg; ldW = CK; }
    else             { W = Wh; ldW = CC; }
    const int c0  = (nt < 2) ? nt * 16 : (nt < 4) ? (nt - 2) * 16 : (nt - 4) * 16;
    const int col = lane & 15;
    const int k   = kstep * 32 + (lane >> 4) * 16 + j;

    Wpack[idx] = (__bf16)W[(size_t)k * ldW + (c0 + col)];
}

// ---------------------------------------------------------------------------
// Kernel 1: projections. One wave = one 16x16 output tile, K=256 unrolled
// (8 WMMAs). A: x rows fp32->bf16 on the fly. B: prepacked bf16 (contiguous).
// Task space: 1024 M-tiles x 20 N-tiles = 20480 waves = 2560 blocks x 8.
// ---------------------------------------------------------------------------
__global__ __launch_bounds__(256)
void proj_kernel(const float* __restrict__ x,
                 const __bf16* __restrict__ Wpack,
                 const float* __restrict__ bfv, const float* __restrict__ bgv,
                 const float* __restrict__ bhv,
                 __bf16* __restrict__ fo, __bf16* __restrict__ go,
                 __bf16* __restrict__ hT)
{
    const int lane = threadIdx.x & 31;
    const int wave = threadIdx.x >> 5;
    const int task = blockIdx.x * 8 + wave;
    const int mt   = task / NT;
    const int nt   = task % NT;
    const int m0   = mt * 16;                  // global pixel row (over B*N)

    const float* bias = (nt < 2) ? bfv : (nt < 4) ? bgv : bhv;
    const int c0 = (nt < 2) ? nt * 16 : (nt < 4) ? (nt - 2) * 16 : (nt - 4) * 16;

    const int col    = lane & 15;
    const int hiHalf = lane >> 4;
    const int h8     = hiHalf * 16;            // A-operand K offset (hi half)
    const int rbase  = hiHalf * 8;             // C/D row offset (hi half)

    const float*  xr = x + (size_t)(m0 + col) * CC;             // A row
    const __bf16* wp = Wpack + ((size_t)nt * 8 * 32 + lane) * 16; // B stream

    v8f acc = {};
    #pragma unroll
    for (int ks = 0; ks < 8; ++ks) {
        const int kk = ks * 32;
        v16bf a;
        #pragma unroll
        for (int j = 0; j < 8; ++j) a[j]     = (__bf16)xr[kk + h8 + j];
        #pragma unroll
        for (int j = 0; j < 8; ++j) a[8 + j] = (__bf16)xr[kk + h8 + 8 + j];
        const v16bf bm = *(const v16bf*)(wp + (size_t)ks * 32 * 16);
        acc = __builtin_amdgcn_wmma_f32_16x16x32_bf16(false, a, false, bm,
                                                      (short)0, acc, false, false);
    }

    const float bv = bias[c0 + col];
    #pragma unroll
    for (int r = 0; r < 8; ++r) acc[r] += bv;

    if (nt < 4) {
        // f/g row-major [B*N, 32]
        __bf16* dst = (nt < 2) ? fo : go;
        #pragma unroll
        for (int r = 0; r < 8; ++r)
            dst[(size_t)(m0 + rbase + r) * CK + (c0 + col)] = (__bf16)acc[r];
    } else {
        // hT [B][C][N]: pixels consecutive -> one 16B store per lane
        const int b  = m0 / NN;
        const int nl = m0 % NN;
        __bf16* dst = hT + ((size_t)b * CC + (c0 + col)) * NN + nl + rbase;
        v8bf pk;
        #pragma unroll
        for (int r = 0; r < 8; ++r) pk[r] = (__bf16)acc[r];
        *(v8bf*)dst = pk;
    }
}

// ---------------------------------------------------------------------------
// Kernel 2: fused flash attention + gamma/residual.
// Grid: 4 batches * 256 query tiles; block: 128 threads = 4 waves.
// Each wave owns 64 output channels (4 f32 accumulators) and redundantly
// computes the cheap S^T tiles (no barriers/LDS). The S^T D-layout feeds the
// PV A-operand with zero cross-lane movement.
// ---------------------------------------------------------------------------
__global__ __launch_bounds__(128)
void attn_kernel(const __bf16* __restrict__ f, const __bf16* __restrict__ g,
                 const __bf16* __restrict__ hT, const float* __restrict__ x,
                 const float* __restrict__ gamma_p, float* __restrict__ y)
{
    const int lane = threadIdx.x & 31;
    const int wave = threadIdx.x >> 5;
    const int b    = blockIdx.x >> 8;
    const int q0   = (blockIdx.x & 255) * 16;
    const int c0   = wave * 64;

    const int col    = lane & 15;
    const int hiHalf = lane >> 4;
    const int h8     = hiHalf * 8;
    const int k16    = hiHalf * 16;
    const int rbase  = hiHalf * 8;

    // G as B-operand for S^T: column = query, K = ck; two 16B chunks (h8, 16+h8) per lane.
    const __bf16* gq = g + ((size_t)b * NN + q0 + col) * CK + h8;
    const v16bf Bg = cat8(*(const v8bf*)gq, *(const v8bf*)(gq + 16));

    float m    = -__builtin_inff();
    float lsum = 0.0f;
    v8f acc0 = {}, acc1 = {}, acc2 = {}, acc3 = {};

    // hoisted streaming pointers
    const __bf16* fp0 = f + (size_t)b * NN * CK + (size_t)col * CK + h8;
    const __bf16* hbase = hT + (size_t)b * CC * NN + h8;
    const __bf16* hp0 = hbase + (size_t)(c0 +  0 + col) * NN;
    const __bf16* hp1 = hbase + (size_t)(c0 + 16 + col) * NN;
    const __bf16* hp2 = hbase + (size_t)(c0 + 32 + col) * NN;
    const __bf16* hp3 = hbase + (size_t)(c0 + 48 + col) * NN;

    for (int kb = 0; kb < NN; kb += 32) {
        // F key tiles as A-operands (two 16B chunks per lane each)
        const __bf16* f0 = fp0 + (size_t)kb * CK;
        const __bf16* f1 = f0 + 16 * CK;
        const v16bf Af0 = cat8(*(const v8bf*)f0, *(const v8bf*)(f0 + 16));
        const v16bf Af1 = cat8(*(const v8bf*)f1, *(const v8bf*)(f1 + 16));

        v8f st0 = {}, st1 = {};
        st0 = __builtin_amdgcn_wmma_f32_16x16x32_bf16(false, Af0, false, Bg,
                                                      (short)0, st0, false, false);
        st1 = __builtin_amdgcn_wmma_f32_16x16x32_bf16(false, Af1, false, Bg,
                                                      (short)0, st1, false, false);

        // ---- online softmax (row = query = lane%16, split over lane^16) ----
        float mloc = st0[0];
        #pragma unroll
        for (int r = 1; r < 8; ++r) mloc = fmaxf(mloc, st0[r]);
        #pragma unroll
        for (int r = 0; r < 8; ++r) mloc = fmaxf(mloc, st1[r]);
        mloc = fmaxf(mloc, __shfl_xor(mloc, 16, 32));
        const float mnew  = fmaxf(m, mloc);
        const float alpha = __expf(m - mnew);     // first iter: exp(-inf)=0
        m = mnew;

        v16bf Ap;                                  // P tile as next A-operand
        float psum = 0.0f;
        #pragma unroll
        for (int r = 0; r < 8; ++r) {
            const float p = __expf(st0[r] - mnew);
            psum += p;  Ap[r] = (__bf16)p;
        }
        #pragma unroll
        for (int r = 0; r < 8; ++r) {
            const float p = __expf(st1[r] - mnew);
            psum += p;  Ap[8 + r] = (__bf16)p;
        }
        psum += __shfl_xor(psum, 16, 32);
        lsum = lsum * alpha + psum;

        // rescale accumulators by per-row alpha (bpermute broadcasts)
        #pragma unroll
        for (int r = 0; r < 8; ++r) {
            const float ar = __shfl(alpha, rbase + r, 32);
            acc0[r] *= ar; acc1[r] *= ar; acc2[r] *= ar; acc3[r] *= ar;
        }

        // ---- P x V: 4 channel tiles, two 16B B-operand chunks per lane ----
        const v16bf Bh0 = cat8(*(const v8bf*)(hp0 + kb), *(const v8bf*)(hp0 + kb + 16));
        const v16bf Bh1 = cat8(*(const v8bf*)(hp1 + kb), *(const v8bf*)(hp1 + kb + 16));
        const v16bf Bh2 = cat8(*(const v8bf*)(hp2 + kb), *(const v8bf*)(hp2 + kb + 16));
        const v16bf Bh3 = cat8(*(const v8bf*)(hp3 + kb), *(const v8bf*)(hp3 + kb + 16));
        acc0 = __builtin_amdgcn_wmma_f32_16x16x32_bf16(false, Ap, false, Bh0,
                                                       (short)0, acc0, false, false);
        acc1 = __builtin_amdgcn_wmma_f32_16x16x32_bf16(false, Ap, false, Bh1,
                                                       (short)0, acc1, false, false);
        acc2 = __builtin_amdgcn_wmma_f32_16x16x32_bf16(false, Ap, false, Bh2,
                                                       (short)0, acc2, false, false);
        acc3 = __builtin_amdgcn_wmma_f32_16x16x32_bf16(false, Ap, false, Bh3,
                                                       (short)0, acc3, false, false);
    }

    // ---- epilogue: y = gamma * (acc / lsum) + x ----
    const float gmm = gamma_p[0];
    #pragma unroll
    for (int r = 0; r < 8; ++r) {
        const float inv = 1.0f / __shfl(lsum, rbase + r, 32);
        const size_t off = ((size_t)b * NN + q0 + rbase + r) * CC;
        const int c = c0 + col;
        y[off + c     ] = gmm * acc0[r] * inv + x[off + c     ];
        y[off + c + 16] = gmm * acc1[r] * inv + x[off + c + 16];
        y[off + c + 32] = gmm * acc2[r] * inv + x[off + c + 32];
        y[off + c + 48] = gmm * acc3[r] * inv + x[off + c + 48];
    }
}

// ---------------------------------------------------------------------------
extern "C" void kernel_launch(void* const* d_in, const int* in_sizes, int n_in,
                              void* d_out, int out_size, void* d_ws, size_t ws_size,
                              hipStream_t stream) {
    const float* x   = (const float*)d_in[0];
    const float* Wf  = (const float*)d_in[1];
    const float* bfv = (const float*)d_in[2];
    const float* Wg  = (const float*)d_in[3];
    const float* bgv = (const float*)d_in[4];
    const float* Wh  = (const float*)d_in[5];
    const float* bhv = (const float*)d_in[6];
    const float* gmm = (const float*)d_in[7];
    float* y = (float*)d_out;

    // workspace: f | g | hT | Wpack   (bf16, ~10.8 MB total)
    __bf16* fo    = (__bf16*)d_ws;
    __bf16* go    = fo + (size_t)BB * NN * CK;
    __bf16* hT    = go + (size_t)BB * NN * CK;
    __bf16* Wpack = hT + (size_t)BB * CC * NN;

    repack_kernel<<<320, 256, 0, stream>>>(Wf, Wg, Wh, Wpack);
    proj_kernel<<<2560, 256, 0, stream>>>(x, Wpack, bfv, bgv, bhv, fo, go, hT);
    attn_kernel<<<1024, 128, 0, stream>>>(fo, go, hT, x, gmm, y);
}
